// MultiHeadedAttentionDecoder_47863115546931
// MI455X (gfx1250) — matrix operand order repaired and checked by reading, not compiled, not run on hardware
//
#include <hip/hip_runtime.h>

// ---------------------------------------------------------------------------
// MultiHeadedAttentionDecoder, MI455X (gfx1250), wave32 + WMMA f16->f32.
// B=16, N=128, D=256, H=8, DK=32.
// ---------------------------------------------------------------------------

typedef __attribute__((ext_vector_type(16))) _Float16 v16h;
typedef __attribute__((ext_vector_type(8)))  _Float16 v8h;
typedef __attribute__((ext_vector_type(4)))  _Float16 v4h;
typedef __attribute__((ext_vector_type(2)))  _Float16 v2h;
typedef __attribute__((ext_vector_type(8)))  float    v8f;

union H16 { v16h v; v8h h[2]; };

constexpr int kB  = 16;
constexpr int kN  = 128;
constexpr int kD  = 256;
constexpr int kH  = 8;
constexpr int kDK = 32;
constexpr int kP  = 264;   // LDS row pitch in halves (528 B, 16B-aligned, bank-skewed)

__device__ __forceinline__ float fast_rcp(float x) {
  return __builtin_amdgcn_rcpf(x);   // v_rcp_f32, 1-ulp, single TRANS op
}

__device__ __forceinline__ float mishf(float x) {
  // mish(x) = x * tanh(softplus(x)) = x * ((1+e^x)^2 - 1) / ((1+e^x)^2 + 1)
  // Use hardware rcp instead of the IEEE divide expansion: this epilogue runs
  // once per edge-output element (67M), so the ~12-op v_div_* sequence would
  // cost as much as the whole memory-bound kernel budget.
  float e  = __expf(x);
  float u  = 1.0f + e;
  float u2 = u * u;
  float t  = (u2 - 1.0f) * fast_rcp(u2 + 1.0f);
  return (x > 20.0f) ? x : x * t;
}

// One 16x16 f32 tile of A(16x256, LDS f16, pitch kP) * B(256x16 from row-major
// f16 weights W[out_col][k]).  klo_a = 8*(lane>=16), klo_b = 16*(lane>=16).
__device__ __forceinline__ v8f wmma_k256(const _Float16* aRow,
                                         const _Float16* bRow,
                                         int klo_a, int klo_b, v8f acc) {
#pragma unroll
  for (int ks = 0; ks < 8; ++ks) {
    H16 a, b;
    const _Float16* ap = aRow + ks * 32 + klo_a;
    a.h[0] = *(const v8h*)(ap);
    a.h[1] = *(const v8h*)(ap + 16);
    const _Float16* bp = bRow + ks * 32 + klo_a;
    b.h[0] = *(const v8h*)(bp);
    b.h[1] = *(const v8h*)(bp + 16);
    acc = __builtin_amdgcn_wmma_f32_16x16x32_f16(
        /*neg_a=*/false, a.v, /*neg_b=*/false, b.v,
        /*c_mod=*/(short)0, acc, /*reuse_a=*/false, /*reuse_b=*/false);
  }
  return acc;
}

// ---------------------------------------------------------------------------
// f32 -> f16 weight conversion (weights stay L2-resident, reused by all WGs)
// ---------------------------------------------------------------------------
__global__ void cvt_f32_to_f16(const float* __restrict__ s,
                               _Float16* __restrict__ d, int n) {
  int i = blockIdx.x * 256 + threadIdx.x;
  if (i < n) d[i] = (_Float16)s[i];
}

// ---------------------------------------------------------------------------
// Generic row GEMM:  Out(rows x 256) = act(X(rows x 256) @ W16^T + bias)
// W16 is row-major f16 [out_col][k].  One block = 32 rows, 8 waves.
// Used for q/v projections (act=0) and the node output (act=1 -> mish).
// ---------------------------------------------------------------------------
__global__ __launch_bounds__(256) void proj_gemm(
    const float* __restrict__ X, const _Float16* __restrict__ W16,
    const float* __restrict__ bias, float* __restrict__ Out, int act) {
  __shared__ _Float16 sx[32 * kP];
  const int tid = threadIdx.x, lane = tid & 31, wave = tid >> 5;
  const size_t row0 = (size_t)blockIdx.x * 32;

  // stage 32 rows of X as f16
  for (int i = tid; i < 32 * 64; i += 256) {
    int r = i >> 6, d4 = (i & 63) << 2;
    float4 f = *(const float4*)(X + (row0 + r) * kD + d4);
    v4h h;
    h[0] = (_Float16)f.x; h[1] = (_Float16)f.y;
    h[2] = (_Float16)f.z; h[3] = (_Float16)f.w;
    *(v4h*)(sx + r * kP + d4) = h;
  }
  __syncthreads();

  const int klo_a = (lane >> 4) * 8, klo_b = (lane >> 4) * 16;
  const int mhi = (lane >> 4) << 3;
  for (int tt = 0; tt < 4; ++tt) {
    int t = wave + (tt << 3);          // 0..31 : 2 M-tiles x 16 N-tiles
    int mt = t & 1, nt = t >> 1;
    int arow = mt * 16 + (lane & 15);
    int bcol = nt * 16 + (lane & 15);
    v8f acc = {};
    acc = wmma_k256(sx + arow * kP, W16 + (size_t)bcol * kD, klo_a, klo_b, acc);
    float bv = bias[bcol];
#pragma unroll
    for (int r = 0; r < 8; ++r) {
      float v = acc[r] + bv;
      if (act) v = mishf(v);
      Out[(row0 + mt * 16 + r + mhi) * kD + bcol] = v;
    }
  }
}

// ---------------------------------------------------------------------------
// Fused per-(b,m) attention kernel.  One WG = 256 threads = 8 waves.
//   GEMM1: Kp = f16(key_edge[b,m]) @ Wk^T + bk          (128x256, WMMA)
//   scores/softmax per head (wave32 shuffles)
//   prot  = attn @ v_proj                               (-> workspace)
//   GEMM2: edge = mish((attn (.) Kp) @ We^T + be)       (128x256, WMMA)
// ---------------------------------------------------------------------------
__global__ __launch_bounds__(256) void fused_attn(
    const float* __restrict__ key_edge, const float* __restrict__ qproj,
    const float* __restrict__ vproj,
    const _Float16* __restrict__ wk16, const float* __restrict__ bk,
    const _Float16* __restrict__ we16, const float* __restrict__ be,
    float* __restrict__ edge_out, float* __restrict__ prot_out) {
  extern __shared__ char smem_raw[];
  _Float16* bufE = (_Float16*)smem_raw;       // 128 x kP halves (E tile / A2)
  _Float16* bufK = bufE + kN * kP;            // 128 x kP halves (Kp)
  float* sc   = (float*)(bufK + kN * kP);     // 8 x 128 scores -> attn
  float* qrow = sc + kH * kN;                 // 256

  const int tid = threadIdx.x, lane = tid & 31, wave = tid >> 5;
  const int bm = blockIdx.x;                  // b*128 + m
  const int b  = bm >> 7;
  const float* Erow = key_edge + (size_t)bm * (kN * kD);

  qrow[tid] = qproj[(size_t)bm * kD + tid];

  // stage key_edge[b,m] (128x256 f32) as f16 into LDS
  for (int i = tid; i < kN * kD / 4; i += 256) {
    int n = i >> 6, d4 = (i & 63) << 2;
    float4 f = *(const float4*)(Erow + n * kD + d4);
    v4h h;
    h[0] = (_Float16)f.x; h[1] = (_Float16)f.y;
    h[2] = (_Float16)f.z; h[3] = (_Float16)f.w;
    *(v4h*)(bufE + n * kP + d4) = h;
  }
  __syncthreads();

  const int klo_a = (lane >> 4) * 8, klo_b = (lane >> 4) * 16;
  const int mhi = (lane >> 4) << 3;

  // ---- GEMM1: Kp = E @ Wk^T + bk, stored f16 in bufK -----------------------
  for (int tt = 0; tt < 16; ++tt) {
    int t = wave * 16 + tt;                 // 0..127 : 8 M-tiles x 16 N-tiles
    int mt = t & 7, nt = t >> 3;
    int arow = mt * 16 + (lane & 15);
    int bcol = nt * 16 + (lane & 15);
    v8f acc = {};
    acc = wmma_k256(bufE + arow * kP, wk16 + (size_t)bcol * kD, klo_a, klo_b, acc);
    float bv = bk[bcol];
#pragma unroll
    for (int r = 0; r < 8; ++r)
      bufK[(mt * 16 + r + mhi) * kP + bcol] = (_Float16)(acc[r] + bv);
  }
  __syncthreads();

  // ---- scores: sc[h][n] = (q[h,:] . Kp[n, h*32:..]) / sqrt(32) -------------
  for (int i = tid; i < kH * kN; i += 256) {
    int h = i >> 7, n = i & 127;
    const _Float16* kp = bufK + n * kP + h * kDK;
    const float* qh = qrow + h * kDK;
    float s = 0.f;
#pragma unroll 8
    for (int d = 0; d < kDK; ++d) s += qh[d] * (float)kp[d];
    sc[i] = s * 0.17677669529663689f;
  }
  __syncthreads();

  // ---- softmax over n, one wave per head -----------------------------------
  {
    float* row = sc + wave * kN;
    float mx = -3.4e38f;
    for (int n = lane; n < kN; n += 32) mx = fmaxf(mx, row[n]);
#pragma unroll
    for (int off = 16; off > 0; off >>= 1) mx = fmaxf(mx, __shfl_xor(mx, off, 32));
    float sum = 0.f;
    for (int n = lane; n < kN; n += 32) {
      float e = __expf(row[n] - mx);
      row[n] = e;
      sum += e;
    }
#pragma unroll
    for (int off = 16; off > 0; off >>= 1) sum += __shfl_xor(sum, off, 32);
    float inv = fast_rcp(sum);
    for (int n = lane; n < kN; n += 32) row[n] *= inv;
  }
  __syncthreads();

  // ---- prot[h][dk] = sum_n attn[h][n] * vproj[b][n][h*32+dk] ---------------
  {
    int h = tid >> 5, dp = tid & 31;
    const float* vb = vproj + (size_t)b * (kN * kD) + h * kDK + dp;
    const float* ar = sc + h * kN;
    float acc = 0.f;
    for (int n = 0; n < kN; ++n) acc += ar[n] * vb[(size_t)n * kD];
    prot_out[(size_t)bm * kD + h * kDK + dp] = acc;
  }

  // ---- A2 = attn (.) Kp, written into bufE (f16) ---------------------------
  for (int i = tid; i < kN * kD / 2; i += 256) {
    int n = i >> 7, d = (i & 127) << 1;     // d even; pair stays in one head
    float a = sc[(d >> 5) * kN + n];
    v2h kv = *(const v2h*)(bufK + n * kP + d);
    v2h ov;
    ov[0] = (_Float16)(a * (float)kv[0]);
    ov[1] = (_Float16)(a * (float)kv[1]);
    *(v2h*)(bufE + n * kP + d) = ov;
  }
  __syncthreads();

  // ---- GEMM2: edge = mish(A2 @ We^T + be) ----------------------------------
  float* orow = edge_out + (size_t)bm * (kN * kD);
  for (int tt = 0; tt < 16; ++tt) {
    int t = wave * 16 + tt;
    int mt = t & 7, nt = t >> 3;
    int arow = mt * 16 + (lane & 15);
    int bcol = nt * 16 + (lane & 15);
    v8f acc = {};
    acc = wmma_k256(bufE + arow * kP, we16 + (size_t)bcol * kD, klo_a, klo_b, acc);
    float bv = be[bcol];
#pragma unroll
    for (int r = 0; r < 8; ++r) {
      float x = acc[r] + bv;
      orow[(size_t)(mt * 16 + r + mhi) * kD + bcol] = mishf(x);
    }
  }
}

// ---------------------------------------------------------------------------
extern "C" void kernel_launch(void* const* d_in, const int* in_sizes, int n_in,
                              void* d_out, int out_size, void* d_ws, size_t ws_size,
                              hipStream_t stream) {
  (void)in_sizes; (void)n_in; (void)out_size; (void)ws_size;

  const float* query = (const float*)d_in[0];
  const float* value = (const float*)d_in[1];
  const float* kedge = (const float*)d_in[2];
  // d_in[3] adj_matrix and d_in[14] attenuation_lambda do not affect outputs.
  const float* Wq = (const float*)d_in[4];
  const float* bq = (const float*)d_in[5];
  const float* Wk = (const float*)d_in[6];
  const float* bk = (const float*)d_in[7];
  const float* Wv = (const float*)d_in[8];
  const float* bv = (const float*)d_in[9];
  const float* We = (const float*)d_in[10];
  const float* be = (const float*)d_in[11];
  const float* Wn = (const float*)d_in[12];
  const float* bn = (const float*)d_in[13];

  // workspace layout
  const int WE = kD * kD;                       // 65536 elems per weight matrix
  _Float16* wq16 = (_Float16*)d_ws;
  _Float16* wk16 = wq16 + WE;
  _Float16* wv16 = wk16 + WE;
  _Float16* we16 = wv16 + WE;
  _Float16* wn16 = we16 + WE;
  float* qproj = (float*)(wn16 + WE);           // B*N*D f32
  float* vproj = qproj + (size_t)kB * kN * kD;  // B*N*D f32
  float* prot  = vproj + (size_t)kB * kN * kD;  // B*N*D f32

  float* edge_out = (float*)d_out;                          // B*N*N*D
  float* node_out = edge_out + (size_t)kB * kN * kN * kD;   // B*N*D

  dim3 blk(256);

  cvt_f32_to_f16<<<WE / 256, blk, 0, stream>>>(Wq, wq16, WE);
  cvt_f32_to_f16<<<WE / 256, blk, 0, stream>>>(Wk, wk16, WE);
  cvt_f32_to_f16<<<WE / 256, blk, 0, stream>>>(Wv, wv16, WE);
  cvt_f32_to_f16<<<WE / 256, blk, 0, stream>>>(We, we16, WE);
  cvt_f32_to_f16<<<WE / 256, blk, 0, stream>>>(Wn, wn16, WE);

  // q / v projections: 2048 rows each, 32 rows per block
  proj_gemm<<<(kB * kN) / 32, blk, 0, stream>>>(query, wq16, bq, qproj, 0);
  proj_gemm<<<(kB * kN) / 32, blk, 0, stream>>>(value, wv16, bv, vproj, 0);

  // fused attention + edge output: one WG per (b,m)
  size_t shmem = (size_t)(2 * kN * kP) * sizeof(_Float16) +
                 (size_t)(kH * kN + kD) * sizeof(float);    // ~140 KB
  hipFuncSetAttribute(reinterpret_cast<const void*>(fused_attn),
                      hipFuncAttributeMaxDynamicSharedMemorySize, (int)shmem);
  fused_attn<<<kB * kN, blk, shmem, stream>>>(kedge, qproj, vproj,
                                              wk16, bk, we16, be,
                                              edge_out, prot);

  // node output: mish(prot @ Wn^T + bn)
  proj_gemm<<<(kB * kN) / 32, blk, 0, stream>>>(prot, wn16, bn, node_out, 1);
}
